// SABase4D_67095979098493
// MI455X (gfx1250) — matrix operand order repaired and checked by reading, not compiled, not run on hardware
//
#include <hip/hip_runtime.h>

// CDNA5 / gfx1250, wave32. Windowed self-attention (q==k) fused kernel.
// Bandwidth-bound (~277MB @ 23.3TB/s ~= 12us; only ~12 GFLOP), so all GEMM
// work goes through v_wmma_f32_16x16x32_f16 with f16 operands staged in LDS,
// f32 accumulation, f32 softmax on the VALU. Global gathers are channel-major
// so consecutive lanes read contiguous pixels (coalesced segments).

typedef _Float16 v16h __attribute__((ext_vector_type(16)));
typedef _Float16 h8   __attribute__((ext_vector_type(8)));
typedef float    v8f  __attribute__((ext_vector_type(8)));

#define DEV __device__ __forceinline__

// Fragment builder for q-derived operands of the score GEMM (K_real = 8).
// A layout (16-bit 16x32): lanes 0-15 row M=lane, e0..7 = K0..7 (K16..23 = 0);
// lanes 16-31 hold K8..15 / K24..31 which are all zero here.
// B layout (32x16): lanes 0-15 col N=lane, e0..7 = K0..7; rest zero.
// Both collapse to the same register image when K_real <= 8.
DEV v16h qfrag(const _Float16* q, int rowbase) {
  int lane = threadIdx.x & 31;
  v16h r = {};
  if (lane < 16) {
    h8 row = *(const h8*)(q + (rowbase + lane) * 8);  // ds_load_b128
#pragma unroll
    for (int i = 0; i < 8; ++i) r[i] = row[i];
  }
  return r;
}

template <int WS, int CH0, int NWAVES>
__global__ __launch_bounds__(NWAVES * 32)
void swin_qv_attn(const float* __restrict__ x, float* __restrict__ out,
                  int nh, int nw) {
  constexpr int H0 = 260, W0 = 260, CPH = 8, HEADS = 4, CTOT = 128;
  constexpr int L     = WS * WS;          // window length: 64 or 256
  constexpr int SHIFT = WS / 2;           // (4,4) / (8,8)
  constexpr int HP = ((H0 + WS - 1) / WS) * WS;  // 264 / 272
  constexpr int WP = ((W0 + WS - 1) / WS) * WS;
  constexpr int NTILE = L / 16;           // 16-row tiles per window
  constexpr int TPW   = NTILE / NWAVES;   // tiles per wave
  constexpr int LSH   = (WS == 8) ? 6 : 8;  // log2(L)
  static_assert(NTILE % NWAVES == 0, "tile split");
  static_assert((1 << LSH) == L, "L must be a power of two");

  extern __shared__ char smem[];
  _Float16* qs = (_Float16*)smem;             // q, row-major  L x 8 f16
  _Float16* vt = (_Float16*)(smem + L * 16);  // v, TRANSPOSED 8 x L f16
  float* Sbase = (float*)(smem + L * 32);     // per-wave 16 x L f32 scores

  const int tid = threadIdx.x, lane = tid & 31, wave = tid >> 5;
  int wj = blockIdx.x % nw;
  int t0 = blockIdx.x / nw;
  int wi = t0 % nh; t0 /= nh;
  int head = t0 % HEADS;
  int bb   = t0 / HEADS;

  // ---- stage q and v^T into LDS (reflect-pad + roll folded into the gather).
  // Channel-major split: consecutive lanes -> consecutive pixels -> contiguous
  // x runs in memory (coalesced), instead of striding across channel planes.
  for (int idx = tid; idx < L * 8; idx += NWAVES * 32) {
    int ch = idx >> LSH;        // 0..7
    int l  = idx & (L - 1);     // pixel index within window
    int Y = wi * WS + (l / WS);
    int X = wj * WS + (l % WS);
    int yp = Y - SHIFT; if (yp < 0) yp += HP;
    if (yp >= H0) yp = 2 * H0 - 2 - yp;     // reflect (high-side pad only)
    int xp = X - SHIFT; if (xp < 0) xp += WP;
    if (xp >= W0) xp = 2 * W0 - 2 - xp;
    long pix = (long)yp * W0 + xp;
    long chq = (long)(bb * CTOT + CH0 + head * CPH + ch);
    long chv = chq + HEADS * CPH;           // v lives 32 channels above q
    float qv = x[chq * (long)(H0 * W0) + pix];
    float vv = x[chv * (long)(H0 * W0) + pix];
    qs[l * 8 + ch] = (_Float16)qv;
    vt[ch * L + l] = (_Float16)vv;
  }
  __syncthreads();

  float* Sw = Sbase + wave * 16 * L;        // this wave's private score strip

  for (int tt = 0; tt < TPW; ++tt) {
    const int tile = wave + tt * NWAVES;

    // ---- scores: S[tile] (16 x L) = q_tile (16x8) . q^T, one WMMA per j-tile
    v16h a = qfrag(qs, tile * 16);
#pragma unroll
    for (int j = 0; j < NTILE; ++j) {
      v16h b = qfrag(qs, j * 16);
      v8f s = {};
      s = __builtin_amdgcn_wmma_f32_16x16x32_f16(false, a, false, b,
                                                 (short)0, s, false, false);
      // C/D layout: lane 0-15 col=lane VGPR p -> M=p; lanes 16-31 -> M=p+8
      int col = (lane & 15) + j * 16;
      int rb  = (lane >> 4) * 8;
#pragma unroll
      for (int p = 0; p < 8; ++p) Sw[(rb + p) * L + col] = s[p];
    }

    // ---- row softmax (f32), emit P as f16 in-place over the score strip.
    // P row r [r*2L, r*2L+2L) only overlaps S rows already consumed (r'<=r).
    _Float16* Pw = (_Float16*)Sw;
    constexpr int EPL = L / 32;             // elements per lane per row
    for (int r = 0; r < 16; ++r) {
      float vals[EPL];
      float m = -3.4e38f;
#pragma unroll
      for (int e = 0; e < EPL; ++e) {
        vals[e] = Sw[r * L + lane + e * 32];
        m = fmaxf(m, vals[e]);
      }
#pragma unroll
      for (int off = 16; off >= 1; off >>= 1)
        m = fmaxf(m, __shfl_xor(m, off, 32));
      float ssum = 0.f;
#pragma unroll
      for (int e = 0; e < EPL; ++e) { vals[e] = __expf(vals[e] - m); ssum += vals[e]; }
#pragma unroll
      for (int off = 16; off >= 1; off >>= 1)
        ssum += __shfl_xor(ssum, off, 32);
      float inv = 1.0f / ssum;
      asm volatile("" ::: "memory");        // keep S reads ahead of aliased P writes
#pragma unroll
      for (int e = 0; e < EPL; ++e)
        Pw[r * L + lane + e * 32] = (_Float16)(vals[e] * inv);
    }

    // ---- o_tile (16x8) = P (16xL) . V (Lx8), K swept in 32-wide chunks
    v8f o = {};
#pragma unroll
    for (int k = 0; k < L / 32; ++k) {
      // A fragment from P: interleaved K layout (K0..7,K16..23 | K8..15,K24..31)
      int row = lane & 15;
      int kab = k * 32 + ((lane & 16) ? 8 : 0);
      v16h pa;
      h8 lo = *(const h8*)(Pw + row * L + kab);
      h8 hi = *(const h8*)(Pw + row * L + kab + 16);
#pragma unroll
      for (int i = 0; i < 8; ++i) { pa[i] = lo[i]; pa[i + 8] = hi[i]; }
      // B fragment from v^T: interleaved K layout (K0..7,K16..23 | K8..15,K24..31)
      v16h vb = {};
      int col = lane & 15;
      int kbb = k * 32 + ((lane & 16) ? 8 : 0);
      if (col < 8) {
        h8 b0 = *(const h8*)(vt + col * L + kbb);
        h8 b1 = *(const h8*)(vt + col * L + kbb + 16);
#pragma unroll
        for (int i = 0; i < 8; ++i) { vb[i] = b0[i]; vb[i + 8] = b1[i]; }
      }
      o = __builtin_amdgcn_wmma_f32_16x16x32_f16(false, pa, false, vb,
                                                 (short)0, o, false, false);
    }

    // ---- writeback with inverse roll + crop (bijective scatter, no races)
    int ch = lane & 15;
    if (ch < 8) {
      int rb = (lane >> 4) * 8;
      long chout = (long)(bb * CTOT + CH0 + head * CPH + ch);
      float* obase = out + chout * (long)(H0 * W0);
#pragma unroll
      for (int p = 0; p < 8; ++p) {
        int l = tile * 16 + rb + p;
        int Y = wi * WS + l / WS, X = wj * WS + l % WS;
        int y  = Y - SHIFT; if (y  < 0) y  += HP;
        int xx = X - SHIFT; if (xx < 0) xx += WP;
        if (y < H0 && xx < W0) obase[(long)y * W0 + xx] = o[p];
      }
    }
    asm volatile("" ::: "memory");          // strip reuse fence for next tile
  }
}

extern "C" void kernel_launch(void* const* d_in, const int* in_sizes, int n_in,
                              void* d_out, int out_size, void* d_ws, size_t ws_size,
                              hipStream_t stream) {
  (void)in_sizes; (void)n_in; (void)d_ws; (void)ws_size; (void)out_size;
  const float* x = (const float*)d_in[0];
  float* out = (float*)d_out;

  // ---- 8x8 windows, shift (4,4): slice channels [0,64), 33x33 windows
  {
    constexpr int SMEM = 64 * 32 + 4 * 16 * 64 * 4;   // qs+vt 2KB, S 16KB = 18.4KB
    (void)hipFuncSetAttribute((const void*)swin_qv_attn<8, 0, 4>,
                              hipFuncAttributeMaxDynamicSharedMemorySize, SMEM);
    dim3 grid(4 * 4 * 33 * 33);                       // b*heads*nh*nw = 17424
    swin_qv_attn<8, 0, 4><<<grid, 128, SMEM, stream>>>(x, out, 33, 33);
  }
  // ---- 16x16 windows, shift (8,8): slice channels [64,128), 17x17 windows
  {
    constexpr int SMEM = 256 * 32 + 8 * 16 * 256 * 4; // 8KB + 128KB = 136KB
    (void)hipFuncSetAttribute((const void*)swin_qv_attn<16, 64, 8>,
                              hipFuncAttributeMaxDynamicSharedMemorySize, SMEM);
    dim3 grid(4 * 4 * 17 * 17);                       // 4624
    swin_qv_attn<16, 64, 8><<<grid, 256, SMEM, stream>>>(x, out, 17, 17);
  }
}
